// RecurrentResonanceModel_76922864271568
// MI455X (gfx1250) — matrix operand order repaired and checked by reading, not compiled, not run on hardware
//
#include <hip/hip_runtime.h>
#include <math.h>

// ---------------------------------------------------------------------------
// RecurrentResonanceModel on gfx1250 (MI455X), wave32 + WMMA f16->f32.
//
// Pipeline (all on `stream`):
//   0) zero_out            : clear d_out (OLA accumulates atomically)
//   1) basis_pack_kernel   : pack windowed iDFT basis [544,512] into f16
//                            B-tile lane order (coalesced b128 loads in GEMM)
//   1b) noise_tr_kernel    : transpose noise [K,NF] -> [NF,K] for coalescing
//   2) mlp_stack_kernel x3 : LinearOutputStack (WMMA 16x16x32 f16) -> ws
//   3) synth_kernel        : 64-frame tile: per-k params once (LDS), closed
//                            form mags/phases -> f16 A panel, then
//                            [64x544]x[544x512] WMMA GEMM (B reused across 4
//                            M-tiles -> 4x less L2 basis traffic), epilogue
//                            50%-hop overlap-add via global_atomic_add_f32.
// ---------------------------------------------------------------------------

typedef _Float16 h2   __attribute__((ext_vector_type(2)));
typedef _Float16 h8   __attribute__((ext_vector_type(8)));
typedef _Float16 h16  __attribute__((ext_vector_type(16)));
typedef float    f8v  __attribute__((ext_vector_type(8)));
typedef float    f4v  __attribute__((ext_vector_type(4)));

#define PI_F 3.14159265358979323846f

namespace rrm {
constexpr int Bn = 8, En = 64, LD = 128, CH = 256;
constexpr int WSZ = 512, RS = 32768, STEP = 256, Kc = 257, NF = 128;
constexpr int ROWS = Bn * En;        // 512 (b,e) rows
constexpr int KK   = 544;            // 2*Kc=514 padded up to multiple of 32
constexpr int NT_B = 32;             // 512/16 output col tiles in synth GEMM
constexpr int KT_B = KK / 32;        // 17 k-steps
constexpr int MROWS = 64;            // frames per synth block (4 M-tiles)
constexpr float BASEv = 0.1f;
constexpr float FACTORv = (1.0f - 0.1f) * 0.999f;   // 0.8991
}

__device__ __forceinline__ float leaky02(float x) { return x > 0.0f ? x : 0.2f * x; }
__device__ __forceinline__ float sigm(float x) { return 1.0f / (1.0f + __expf(-x)); }

// ---------------------------------------------------------------------------
// Wave-level GEMM tile helper (MLP).  A: f16 in LDS [16 x Kdim] (row stride
// astr, 16B-aligned rows).  B: f32 weights in global, row-major [Kdim x ldw],
// converted to f16 on the fly.  Output: C[16 x nTiles*16] f32 either to an
// LDS activation buffer (stride 256) or to global rows [rowBase.. , gld].
// A/B use contiguous k order (k = 16*(lane/16) + s) in both; f32 C/D 16x16 per CDNA5 ISA 7.12.2.
// ---------------------------------------------------------------------------
__device__ __forceinline__ void wave_gemm(
    const _Float16* __restrict__ A, int astr,
    const float* __restrict__ W, int ldw, int Kdim,
    int nTiles, int Nvalid, const float* __restrict__ bias,
    float* __restrict__ actOut,              // LDS, stride 256 (or nullptr)
    float* __restrict__ gOut, int gld, int rowBase)
{
  const int lane  = threadIdx.x & 31;
  const int wave  = threadIdx.x >> 5;
  const int nwave = blockDim.x >> 5;
  const int lmod  = lane & 15;
  const int lhi   = lane >> 4;                 // 0 or 1

  for (int nt = wave; nt < nTiles; nt += nwave) {
    f8v acc = {0.f, 0.f, 0.f, 0.f, 0.f, 0.f, 0.f, 0.f};
    const int n  = nt * 16 + lmod;
    const int nl = (n < Nvalid) ? n : (Nvalid - 1);   // clamp OOB loads

    const int ksteps = Kdim >> 5;
    for (int kt = 0; kt < ksteps; ++kt) {
      // A tile (16x32 f16): lanes 0-15 hold K 0-15, lanes 16-31 hold
      // K 16-31 -> two contiguous 16B LDS loads.
      const int k0 = (kt << 5) + (lhi << 4);
      const _Float16* ap = A + lmod * astr + k0;
      h8 alo = *(const h8*)ap;
      h8 ahi = *(const h8*)(ap + 8);
      h16 a;
#pragma unroll
      for (int i = 0; i < 8; ++i) { a[i] = alo[i]; a[i + 8] = ahi[i]; }

      // B tile (32x16): lane holds column n, halves j -> K = kb+j.
      const int kb = (kt << 5) + (lhi << 4);
      h16 bm;
#pragma unroll
      for (int j = 0; j < 16; ++j)
        bm[j] = (_Float16)W[(size_t)(kb + j) * (size_t)ldw + nl];

      acc = __builtin_amdgcn_wmma_f32_16x16x32_f16(
          false, a, false, bm, (short)0, acc, false, false);
    }

    const float bv = bias[nl];
#pragma unroll
    for (int v = 0; v < 8; ++v) {
      const int m = v + (lhi << 3);            // C/D layout: row = v + 8*hi
      const float val = acc[v] + bv;
      if (actOut) {
        actOut[m * 256 + n] = val;             // hidden: Nvalid==256 exactly
      } else if (n < Nvalid) {
        gOut[(size_t)(rowBase + m) * (size_t)gld + n] = val;
      }
    }
  }
}

// ---------------------------------------------------------------------------
// LinearOutputStack: one block = one 16-row tile (128 threads / 4 waves).
//   x@Wi+bi -> 3x[leaky->LN->@Wh+bh] -> leaky -> @Wo+bo   -> Yout[512x257]
// ---------------------------------------------------------------------------
__global__ void __launch_bounds__(128)
mlp_stack_kernel(const float* __restrict__ X,            // [512,128]
                 const float* __restrict__ Wi, const float* __restrict__ bi,
                 const float* __restrict__ Wh, const float* __restrict__ bh,
                 const float* __restrict__ g,  const float* __restrict__ be,
                 const float* __restrict__ Wo, const float* __restrict__ bo,
                 float* __restrict__ Yout)                // [512,257]
{
  using namespace rrm;
  __shared__ __align__(16) _Float16 Ain[16][288];   // f16 A panel (K<=256)
  __shared__ float act[16][256];                    // f32 activations
  __shared__ float rmean[16], rinv[16];

  const int rowBase = blockIdx.x * 16;
  const int tid = threadIdx.x;

  // input -> f16 A panel
  for (int idx = tid; idx < 16 * 128; idx += 128) {
    const int m = idx >> 7, k = idx & 127;
    Ain[m][k] = (_Float16)X[(size_t)(rowBase + m) * 128 + k];
  }
  __syncthreads();

  // input projection: [16,128]x[128,256]
  wave_gemm(&Ain[0][0], 288, Wi, CH, 128, 16, 256, bi, &act[0][0], nullptr, 0, 0);
  __syncthreads();

  for (int l = 0; l < 3; ++l) {
    // leaky_relu in place
    for (int idx = tid; idx < 16 * 256; idx += 128) {
      const int m = idx >> 8, k = idx & 255;
      act[m][k] = leaky02(act[m][k]);
    }
    __syncthreads();
    // LayerNorm stats (one thread per row; 256 elems each — tiny)
    if (tid < 16) {
      float s = 0.f, s2 = 0.f;
      for (int k = 0; k < 256; ++k) { const float v = act[tid][k]; s += v; s2 += v * v; }
      const float m = s * (1.0f / 256.0f);
      const float var = s2 * (1.0f / 256.0f) - m * m;
      rmean[tid] = m;
      rinv[tid]  = rsqrtf(var + 1e-5f);
    }
    __syncthreads();
    // normalize * g + be  -> f16 A panel
    for (int idx = tid; idx < 16 * 256; idx += 128) {
      const int m = idx >> 8, k = idx & 255;
      const float v = (act[m][k] - rmean[m]) * rinv[m] * g[l * 256 + k] + be[l * 256 + k];
      Ain[m][k] = (_Float16)v;
    }
    __syncthreads();
    // hidden linear: [16,256]x[256,256]
    wave_gemm(&Ain[0][0], 288, Wh + (size_t)l * 256 * 256, CH, 256, 16, 256,
              bh + l * 256, &act[0][0], nullptr, 0, 0);
    __syncthreads();
  }

  // final leaky -> f16, then out projection [16,256]x[256,257] (17 tiles)
  for (int idx = tid; idx < 16 * 256; idx += 128) {
    const int m = idx >> 8, k = idx & 255;
    Ain[m][k] = (_Float16)leaky02(act[m][k]);
  }
  __syncthreads();
  wave_gemm(&Ain[0][0], 288, Wo, Kc, 256, 17, Kc, bo, nullptr, Yout, Kc, rowBase);
}

// ---------------------------------------------------------------------------
// Pack the windowed ortho-iDFT basis (window + 1/sqrt(N) + DC/Nyquist weights
// folded in) as f16 in B-tile lane order:
//   Bpack[(((kt*32+nt)*32 + lane)*16 + j)] = basis[kk = kt*32+(lane/16)*16+j]
//                                                 [ t = nt*16 + lane%16 ]
// kk even -> +c*cos, kk odd -> -c*sin (frames stored Re/Im interleaved).
// ---------------------------------------------------------------------------
__global__ void basis_pack_kernel(_Float16* __restrict__ Bpack)
{
  using namespace rrm;
  const int idx = blockIdx.x * blockDim.x + threadIdx.x;  // [0, 17*32*32*16)
  const int j    = idx & 15;
  const int lane = (idx >> 4) & 31;
  const int ktnt = idx >> 9;
  const int nt   = ktnt & 31;
  const int kt   = ktnt >> 5;
  const int kk = kt * 32 + ((lane >> 4) << 4) + j;
  const int t  = nt * 16 + (lane & 15);
  float val = 0.0f;
  if (kk < 2 * Kc) {
    const int k = kk >> 1;
    const int ktm = (k * t) & (WSZ - 1);               // exact mod-2pi reduction
    const float ang = (float)ktm * (2.0f * PI_F / (float)WSZ);
    const float ak = (k == 0 || k == WSZ / 2) ? 1.0f : 2.0f;
    const float win = 0.5f - 0.5f * cosf((float)t * (2.0f * PI_F / (float)WSZ));
    const float c = ak * win * (1.0f / 22.62741699796952f);   // 1/sqrt(512)
    val = (kk & 1) ? (-c * sinf(ang)) : (c * cosf(ang));
  }
  Bpack[idx] = (_Float16)val;
}

// transpose noise [K=257, NF=128] -> noiseT [NF=128, K=257]
__global__ void noise_tr_kernel(const float* __restrict__ noise,
                                float* __restrict__ noiseT)
{
  using namespace rrm;
  const int idx = blockIdx.x * blockDim.x + threadIdx.x;
  if (idx < Kc * NF) {
    const int k = idx / NF, f = idx - k * NF;
    noiseT[f * Kc + k] = noise[idx];
  }
}

// ---------------------------------------------------------------------------
// Synth: one block = 64 consecutive frames of one (b,e) row (256 thr/8 waves).
// Phase 0: per-k params (init, log2(res), dither) once per block -> LDS.
// Phase 1: closed-form frame spectra -> f16 A panel [64 x 544] in LDS.
// Phase 2: WMMA GEMM [64x544]x[544x512]; wave w owns 4 N-tiles, loads B once
//          per k-step and reuses it across the 4 M-tiles (16 wmma / 8 loads).
// Phase 3: epilogue overlap-add via atomicAdd into d_out.
// ---------------------------------------------------------------------------
__global__ void __launch_bounds__(256)
synth_kernel(const float* __restrict__ initO,   // [512,257]
             const float* __restrict__ resO,    // [512,257]
             const float* __restrict__ dthO,    // [512,257]
             const float* __restrict__ noiseT,  // [128,257]
             const _Float16* __restrict__ Bpack,
             float* __restrict__ out)           // [512,32768]
{
  using namespace rrm;
  __shared__ __align__(16) _Float16 Ain[MROWS][KK];   // 69.6 KB
  __shared__ float ivS[Kc], l2S[Kc], dvS[Kc];

  const int r0 = blockIdx.x * MROWS;
  const int be = r0 >> 7;          // constant within tile (64 | 128)
  const int f0 = r0 & 127;
  const int tid = threadIdx.x;

  // ---- Phase 0: per-k parameters (once per block)
  for (int k = tid; k < Kc; k += 256) {
    const float gd = (float)k * (PI_F / 256.0f);
    ivS[k] = initO[(size_t)be * Kc + k];
    l2S[k] = __log2f(BASEv + sigm(resO[(size_t)be * Kc + k]) * FACTORv);
    dvS[k] = sigm(dthO[(size_t)be * Kc + k]) * gd;
  }
  __syncthreads();

  // ---- Phase 1: mags/phases in closed form -> Re/Im interleaved f16 panel
  for (int idx = tid; idx < MROWS * 272; idx += 256) {
    const int m  = idx / 272;
    const int kq = idx - m * 272;
    h2 pair = {(_Float16)0.0f, (_Float16)0.0f};
    int k = kq;
    if (kq < Kc) {
      const int f = f0 + m;
      const float fp1 = (float)(f + 1);
      const float mag = ivS[k] * exp2f(fp1 * l2S[k]);
      const int kf = (k * (f + 1)) & (WSZ - 1);        // gd*(f+1) mod 2pi
      const float ph = (float)kf * (PI_F / 256.0f) + noiseT[f * Kc + k] * dvS[k];
      float s, c;
      __sincosf(ph, &s, &c);
      pair[0] = (_Float16)(mag * c);
      pair[1] = (_Float16)(mag * s);
    }
    *(h2*)&Ain[m][2 * k] = pair;                        // one 32-bit ds store
  }
  __syncthreads();

  // ---- Phase 2: GEMM. wave w owns N-tiles nt = w*4+j, all 4 M-tiles.
  const int lane = tid & 31;
  const int wave = tid >> 5;
  const int lmod = lane & 15;
  const int lhi  = lane >> 4;

  f8v acc[4][4];                                        // [mt][j]
#pragma unroll
  for (int mt = 0; mt < 4; ++mt)
#pragma unroll
    for (int j = 0; j < 4; ++j)
      acc[mt][j] = (f8v){0.f,0.f,0.f,0.f,0.f,0.f,0.f,0.f};

  for (int kt = 0; kt < KT_B; ++kt) {
    // load this wave's 4 B tiles once (2 x b128 each, coalesced)
    h16 b[4];
#pragma unroll
    for (int j = 0; j < 4; ++j) {
      const int nt = wave * 4 + j;
      const _Float16* bp = Bpack + ((size_t)((kt * 32 + nt) * 32 + lane) << 4);
      h8 blo = *(const h8*)bp;
      h8 bhi = *(const h8*)(bp + 8);
#pragma unroll
      for (int i = 0; i < 8; ++i) { b[j][i] = blo[i]; b[j][i + 8] = bhi[i]; }
    }
    // for each M-tile: load A once, feed all 4 B tiles (16 wmma / k-step)
    const int k0 = (kt << 5) + (lhi << 4);
#pragma unroll
    for (int mt = 0; mt < 4; ++mt) {
      const _Float16* ap = &Ain[mt * 16 + lmod][k0];
      h8 alo = *(const h8*)ap;
      h8 ahi = *(const h8*)(ap + 8);
      h16 a;
#pragma unroll
      for (int i = 0; i < 8; ++i) { a[i] = alo[i]; a[i + 8] = ahi[i]; }
#pragma unroll
      for (int j = 0; j < 4; ++j)
        acc[mt][j] = __builtin_amdgcn_wmma_f32_16x16x32_f16(
            false, a, false, b[j], (short)0, acc[mt][j], false, false);
    }
  }

  // ---- Phase 3: 50%-hop overlap-add (two atomic contributions per sample)
  float* orow = out + (size_t)be * RS;
#pragma unroll
  for (int mt = 0; mt < 4; ++mt) {
#pragma unroll
    for (int j = 0; j < 4; ++j) {
      const int nt = wave * 4 + j;
      const int t = nt * 16 + lmod;
#pragma unroll
      for (int v = 0; v < 8; ++v) {
        const int m = mt * 16 + v + (lhi << 3);
        const int f = f0 + m;
        const float val = acc[mt][j][v];
        if (t < STEP) {
          atomicAdd(&orow[f * STEP + t], val);
        } else {
          const int s = (f + 1) * STEP + (t - STEP);
          if (s < RS) atomicAdd(&orow[s], val);
        }
      }
    }
  }
}

__global__ void zero_out_kernel(float* __restrict__ out, int n4)
{
  f4v z = {0.f, 0.f, 0.f, 0.f};
  for (int i = blockIdx.x * blockDim.x + threadIdx.x; i < n4;
       i += gridDim.x * blockDim.x)
    ((f4v*)out)[i] = z;
}

// ---------------------------------------------------------------------------
extern "C" void kernel_launch(void* const* d_in, const int* in_sizes, int n_in,
                              void* d_out, int out_size, void* d_ws, size_t ws_size,
                              hipStream_t stream) {
  using namespace rrm;
  (void)in_sizes; (void)n_in; (void)ws_size;

  const float* latents = (const float*)d_in[0];
  const float* noise   = (const float*)d_in[1];

  // per-stack params: Wi, bi, Wh, bh, g, be, Wo, bo at d_in[2 + 8*s .. ]
  const float* P[3][8];
  for (int s = 0; s < 3; ++s)
    for (int p = 0; p < 8; ++p)
      P[s][p] = (const float*)d_in[2 + s * 8 + p];

  float* outF = (float*)d_out;

  // workspace layout: 3 x [512*257] f32 stack outputs, basis f16, noiseT f32
  const size_t stackElems = (size_t)ROWS * Kc;               // 131584
  float* initOut = (float*)d_ws;
  float* resOut  = initOut + stackElems;
  float* dthOut  = resOut + stackElems;
  _Float16* Bpack = (_Float16*)(dthOut + stackElems);
  float* noiseT = (float*)(Bpack + (size_t)KT_B * NT_B * 32 * 16);

  // 0) zero output (overlap-add accumulates atomically)
  zero_out_kernel<<<4096, 256, 0, stream>>>(outF, out_size / 4);

  // 1) pack windowed iDFT basis: 17*32*32*16 = 278528 f16 elements
  basis_pack_kernel<<<(KT_B * NT_B * 32 * 16) / 256, 256, 0, stream>>>(Bpack);

  // 1b) transpose noise for coalesced phase-1 reads
  noise_tr_kernel<<<(Kc * NF + 255) / 256, 256, 0, stream>>>(noise, noiseT);

  // 2) three LinearOutputStacks (32 row-tiles each)
  float* outs[3] = {initOut, resOut, dthOut};
  for (int s = 0; s < 3; ++s) {
    mlp_stack_kernel<<<ROWS / 16, 128, 0, stream>>>(
        latents, P[s][0], P[s][1], P[s][2], P[s][3],
        P[s][4], P[s][5], P[s][6], P[s][7], outs[s]);
  }

  // 3) synth + iDFT GEMM + OLA: 65536 frame-rows / 64 per block
  synth_kernel<<<(ROWS * NF) / MROWS, 256, 0, stream>>>(
      initOut, resOut, dthOut, noiseT, Bpack, outF);
}
